// SelfAttention_35656818492218
// MI455X (gfx1250) — matrix operand order repaired and checked by reading, not compiled, not run on hardware
//
#include <hip/hip_runtime.h>

#define B_   8
#define C_   512
#define N_   2048
#define CQK_ 64

typedef __bf16 bf16;
typedef __attribute__((ext_vector_type(16))) __bf16 v16bf;
typedef __attribute__((ext_vector_type(8)))  __bf16 v8bf;
typedef __attribute__((ext_vector_type(8)))  float  v8f;
typedef unsigned int u32;
typedef __attribute__((ext_vector_type(4))) u32 v4u;
typedef __attribute__((ext_vector_type(8))) int v8i;
typedef __attribute__((ext_vector_type(4))) int v4i;

#define WMMA_BF16(a, b, c) \
  __builtin_amdgcn_wmma_f32_16x16x32_bf16(false, (a), false, (b), (short)0, (c), false, false)

// A-operand fragment (16M x 32K bf16): lane M = lane&15, half selects K groups.
// e<8 -> K = half*8+e ; e>=8 -> K = 16+half*8+(e-8). Two contiguous 16B loads.
static __device__ __forceinline__ v16bf load_A(const bf16* __restrict__ row, int k0, int half) {
  v8bf lo = *(const v8bf*)(row + k0 + half * 8);
  v8bf hi = *(const v8bf*)(row + k0 + 16 + half * 8);
  v16bf r;
#pragma unroll
  for (int e = 0; e < 8; ++e) { r[e] = lo[e]; r[e + 8] = hi[e]; }
  return r;
}

// B-operand fragment (32K x 16N bf16): lane n = lane&15, e<8 -> K = half*8+e ; e>=8 -> K = 16+half*8+(e-8).
static __device__ __forceinline__ v16bf load_B(const bf16* __restrict__ row, int k0, int half) {
  v8bf lo = *(const v8bf*)(row + k0 + half * 8);
  v8bf hi = *(const v8bf*)(row + k0 + 16 + half * 8);
  v16bf r;
#pragma unroll
  for (int e = 0; e < 8; ++e) { r[e] = lo[e]; r[e + 8] = hi[e]; }
  return r;
}

// ---- Tensor Data Mover: flat 1-D copy of `units8` 8-byte units global -> LDS ----
// D# per CDNA5 ISA 8.3/8.4: group0 {count=1, lds_addr, global_addr, type=2},
// group1 {data_size=3 (8B), tensor_dim0=tile_dim0=stride0=units8, tile_dim1=1}.
// This toolchain exposes the 6-arg builtin (g0, g1, g2, g3, g4, cpol).
static __device__ __forceinline__ void tdm_load_to_lds(u32 lds_off, const void* gptr, u32 units8) {
  u32 ga_lo = (u32)(uintptr_t)gptr;
  u32 ga_hi = (u32)((uintptr_t)gptr >> 32);
  v4u g0 = { 1u,                                   // [1:0] count=1
             lds_off,                              // [63:32] lds_addr
             ga_lo,                                // [95:64] global_addr lo
             (ga_hi & 0x01FFFFFFu) | (2u << 30) }; // [120:96] addr hi, [127:126] type=2
  v8i g1;
  g1[0] = (int)(3u << 16);        // [15:0] wg_mask=0, [17:16] data_size=3 (8B)
  g1[1] = (int)(units8 << 16);    // [79:48] tensor_dim0 (low 16)
  g1[2] = (int)(1u << 16);        // [79:64] dim0 hi=0, [111:80] tensor_dim1=1 (low 16)
  g1[3] = (int)(units8 << 16);    // [111:96] dim1 hi=0, [127:112] tile_dim0
  g1[4] = 1;                      // [143:128] tile_dim1=1, [159:144] tile_dim2=0
  g1[5] = (int)units8;            // [207:160] tensor_dim0_stride (low 32)
  g1[6] = 0;                      // stride hi, tensor_dim1_stride lo
  g1[7] = 0;
  v4i z4 = { 0, 0, 0, 0 };
  v8i z8 = { 0, 0, 0, 0, 0, 0, 0, 0 };
  __builtin_amdgcn_tensor_load_to_lds(g0, g1, z4, z4, z8, 0);
}

// ---------------- elementwise f32 -> bf16 ----------------
__global__ void cvt_bf16_kernel(const float* __restrict__ in, bf16* __restrict__ out, int n) {
  int i = blockIdx.x * blockDim.x + threadIdx.x;
  if (i < n) out[i] = (bf16)in[i];
}

// ---------------- x [B][C][N] f32 -> xT [B][N][C] bf16 ----------------
__global__ void transpose_x_kernel(const float* __restrict__ x, bf16* __restrict__ xT) {
  __shared__ float t[32][33];
  const int b = blockIdx.z;
  const int n0 = blockIdx.x * 32, c0 = blockIdx.y * 32;
  const int tx = threadIdx.x, ty = threadIdx.y;
#pragma unroll
  for (int k = 0; k < 4; ++k)
    t[ty + k * 8][tx] = x[((size_t)b * C_ + c0 + ty + k * 8) * N_ + n0 + tx];
  __syncthreads();
#pragma unroll
  for (int k = 0; k < 4; ++k)
    xT[((size_t)b * N_ + n0 + ty + k * 8) * C_ + c0 + tx] = (bf16)t[tx][ty + k * 8];
}

// ---------------- projection GEMM: out = W(MxC) * x(CxN) + bias ----------------
__global__ void proj_kernel(const bf16* __restrict__ W, const float* __restrict__ bias,
                            const bf16* __restrict__ xT, bf16* __restrict__ out,
                            int M, int transposedStore) {
  const int lane = threadIdx.x & 31, wave = threadIdx.x >> 5;
  const int half = lane >> 4, ln = lane & 15;
  const int tilesM = M >> 4, tilesN = N_ >> 4;
  const int perB = tilesM * tilesN;
  int gid = blockIdx.x * (blockDim.x >> 5) + wave;
  int b = gid / perB;
  if (b >= B_) return;  // wave-uniform
  int rem = gid - b * perB;
  int mt = rem / tilesN, nt = rem - mt * tilesN;
  const int m0 = mt << 4, n0 = nt << 4;

  const bf16* arow = W + (size_t)(m0 + ln) * C_;
  const bf16* brow = xT + ((size_t)b * N_ + n0 + ln) * C_;
  v8f acc = {};
#pragma unroll 4
  for (int k0 = 0; k0 < C_; k0 += 32)
    acc = WMMA_BF16(load_A(arow, k0, half), load_B(brow, k0, half), acc);

  if (transposedStore) {  // Qt/Kt: [B][N][CQK]; lane's 8 D rows contiguous -> one b128
    v8bf pv;
#pragma unroll
    for (int r = 0; r < 8; ++r) pv[r] = (bf16)(acc[r] + bias[m0 + 8 * half + r]);
    *(v8bf*)(out + ((size_t)b * N_ + n0 + ln) * CQK_ + m0 + 8 * half) = pv;
  } else {               // V: natural [B][M][N]
#pragma unroll
    for (int r = 0; r < 8; ++r) {
      int m = m0 + r + 8 * half;
      out[((size_t)b * M + m) * N_ + n0 + ln] = (bf16)(acc[r] + bias[m]);
    }
  }
}

// ---------------- softmax row stats: m_i, 1/Z_i over j ----------------
__global__ void stats_kernel(const bf16* __restrict__ Qt, const bf16* __restrict__ Kt,
                             float* __restrict__ mOut, float* __restrict__ zinvOut) {
  const int lane = threadIdx.x & 31, wave = threadIdx.x >> 5;
  const int half = lane >> 4, ln = lane & 15;
  const int b = blockIdx.x >> 7;
  const int i0 = (blockIdx.x & 127) << 4;

  const bf16* arow = Qt + ((size_t)b * N_ + i0 + ln) * CQK_;
  v16bf a0 = load_A(arow, 0, half);
  v16bf a1 = load_A(arow, 32, half);

  float mloc[8], zloc[8];
#pragma unroll
  for (int r = 0; r < 8; ++r) { mloc[r] = -3.0e38f; zloc[r] = 0.f; }

  for (int jt = wave; jt < (N_ >> 4); jt += 4) {
    const bf16* brow = Kt + ((size_t)b * N_ + (jt << 4) + ln) * CQK_;
    v8f acc = {};
    acc = WMMA_BF16(a0, load_B(brow, 0, half), acc);
    acc = WMMA_BF16(a1, load_B(brow, 32, half), acc);
#pragma unroll
    for (int r = 0; r < 8; ++r) {
      float e = acc[r];
      float mn = fmaxf(mloc[r], e);
      zloc[r] = zloc[r] * __expf(mloc[r] - mn) + __expf(e - mn);
      mloc[r] = mn;
    }
  }
#pragma unroll
  for (int mask = 8; mask >= 1; mask >>= 1) {
#pragma unroll
    for (int r = 0; r < 8; ++r) {
      float mo = __shfl_xor(mloc[r], mask, 16);
      float zo = __shfl_xor(zloc[r], mask, 16);
      float mn = fmaxf(mloc[r], mo);
      zloc[r] = zloc[r] * __expf(mloc[r] - mn) + zo * __expf(mo - mn);
      mloc[r] = mn;
    }
  }
  __shared__ float sm[4][16], sz[4][16];
  if (ln == 0) {
#pragma unroll
    for (int r = 0; r < 8; ++r) { sm[wave][r + 8 * half] = mloc[r]; sz[wave][r + 8 * half] = zloc[r]; }
  }
  __syncthreads();
  if (threadIdx.x < 16) {
    float m = sm[0][threadIdx.x], z = sz[0][threadIdx.x];
#pragma unroll
    for (int w = 1; w < 4; ++w) {
      float mo = sm[w][threadIdx.x], zo = sz[w][threadIdx.x];
      float mn = fmaxf(m, mo);
      z = z * __expf(m - mn) + zo * __expf(mo - mn);
      m = mn;
    }
    mOut[(size_t)b * N_ + i0 + threadIdx.x] = m;
    zinvOut[(size_t)b * N_ + i0 + threadIdx.x] = 1.0f / z;
  }
}

// ---------------- Vs[c,i] = V[c,i] * zinv[i] ----------------
__global__ void scalev_kernel(const bf16* __restrict__ V, const float* __restrict__ zinv,
                              bf16* __restrict__ Vs) {
  size_t idx = (size_t)blockIdx.x * blockDim.x + threadIdx.x;
  if (idx >= (size_t)B_ * C_ * N_) return;
  int i = (int)(idx % N_);
  int b = (int)(idx / ((size_t)C_ * N_));
  Vs[idx] = (bf16)((float)V[idx] * zinv[(size_t)b * N_ + i]);
}

// ---------------- fused attention output: out = gamma * Vs * exp(QtKt - m) + x ----------------
// block = 16 waves; (batch b, 64 j-cols, all 512 c-rows).
// TDM stages Kt j-block once and double-buffers Qt i-step tiles into LDS (TENSORcnt-pipelined).
// Phase 1: waves 0-7 build P^T [64j][32i] in LDS via WMMA + exp from LDS operands.
// Phase 2: all 16 waves accumulate 32 c-rows x 64 j with 8 WMMAs per i-step.
__global__ void __launch_bounds__(512) attn_out_kernel(
    const bf16* __restrict__ Qt, const bf16* __restrict__ Kt,
    const bf16* __restrict__ Vs, const float* __restrict__ mRow,
    const float* __restrict__ x, const float* __restrict__ gamma,
    float* __restrict__ out) {
  const int lane = threadIdx.x & 31, wave = threadIdx.x >> 5;
  const int half = lane >> 4, ln = lane & 15;
  const int b = blockIdx.x >> 5;
  const int j0 = (blockIdx.x & 31) << 6;

  __shared__ __align__(16) bf16 Pt[64][32];        // P^T [j_local][i_local]   4KB
  __shared__ __align__(16) bf16 sKt[64][64];       // Kt j-block               8KB
  __shared__ __align__(16) bf16 sQt[2][32][64];    // Qt i-step double buffer  8KB
  __shared__ float sM[N_];                         // row max for this batch   8KB

  // stage mRow[b][*] into LDS (cooperative, once)
  for (int t = threadIdx.x; t < N_; t += 512) sM[t] = mRow[(size_t)b * N_ + t];

  // TDM prologue: Kt block (8KB) + first Qt tile (4KB); contiguous in our layouts
  if (wave == 0) {
    tdm_load_to_lds((u32)(uintptr_t)&sKt[0][0], Kt + ((size_t)b * N_ + j0) * CQK_, 1024);
    tdm_load_to_lds((u32)(uintptr_t)&sQt[0][0][0], Qt + (size_t)b * N_ * CQK_, 512);
  }

  v8f zero = {};
  v8f acc[2][4];
#pragma unroll
  for (int ct = 0; ct < 2; ++ct)
#pragma unroll
    for (int jt = 0; jt < 4; ++jt) acc[ct][jt] = zero;

  const int c0 = wave * 32;
  const bf16* va0row = Vs + ((size_t)b * C_ + c0 + ln) * N_;
  const bf16* va1row = Vs + ((size_t)b * C_ + c0 + 16 + ln) * N_;

  for (int k = 0; k < N_ / 32; ++k) {
    const int i0 = k * 32, cur = k & 1;
    if (wave == 0) {
      if (k + 1 < N_ / 32) {  // prefetch next Qt tile into other buffer
        tdm_load_to_lds((u32)(uintptr_t)&sQt[cur ^ 1][0][0],
                        Qt + ((size_t)b * N_ + i0 + 32) * CQK_, 512);
        __builtin_amdgcn_s_wait_tensorcnt(1);  // current tile landed, prefetch in flight
      } else {
        __builtin_amdgcn_s_wait_tensorcnt(0);
      }
    }
    __syncthreads();  // publish sQt[cur] (and guard Pt against prev-iter readers)

    if (wave < 8) {   // phase 1: energy -> P^T tile
      const int ti = wave >> 2, tj = wave & 3;
      const bf16* arow = &sQt[cur][ti * 16 + ln][0];
      const bf16* brow = &sKt[tj * 16 + ln][0];
      v8f e = zero;
      e = WMMA_BF16(load_A(arow, 0, half), load_B(brow, 0, half), e);
      e = WMMA_BF16(load_A(arow, 32, half), load_B(brow, 32, half), e);
      v8bf pv;
#pragma unroll
      for (int r = 0; r < 8; ++r)
        pv[r] = (bf16)__expf(e[r] - sM[i0 + ti * 16 + r + 8 * half]);
      *(v8bf*)(&Pt[tj * 16 + ln][ti * 16 + 8 * half]) = pv;  // contiguous 16B LDS store
    }
    __syncthreads();

    // phase 2: out-accumulate
    v16bf va0 = load_A(va0row, i0, half);
    v16bf va1 = load_A(va1row, i0, half);
#pragma unroll
    for (int jt = 0; jt < 4; ++jt) {
      v16bf vb = load_B(&Pt[jt * 16 + ln][0], 0, half);
      acc[0][jt] = WMMA_BF16(va0, vb, acc[0][jt]);
      acc[1][jt] = WMMA_BF16(va1, vb, acc[1][jt]);
    }
  }

  const float g = gamma[0];
#pragma unroll
  for (int ct = 0; ct < 2; ++ct)
#pragma unroll
    for (int jt = 0; jt < 4; ++jt)
#pragma unroll
      for (int r = 0; r < 8; ++r) {
        int c = c0 + ct * 16 + r + 8 * half;
        int j = j0 + jt * 16 + ln;
        size_t idx = ((size_t)b * C_ + c) * N_ + j;
        out[idx] = g * acc[ct][jt][r] + x[idx];
      }
}

extern "C" void kernel_launch(void* const* d_in, const int* in_sizes, int n_in,
                              void* d_out, int out_size, void* d_ws, size_t ws_size,
                              hipStream_t stream) {
  const float* x     = (const float*)d_in[0];
  const float* Wq    = (const float*)d_in[1];
  const float* bq    = (const float*)d_in[2];
  const float* Wk    = (const float*)d_in[3];
  const float* bk    = (const float*)d_in[4];
  const float* Wv    = (const float*)d_in[5];
  const float* bv    = (const float*)d_in[6];
  const float* gamma = (const float*)d_in[7];
  float* out = (float*)d_out;

  char* ws = (char*)d_ws;
  size_t off = 0;
  auto take = [&](size_t bytes) -> char* {
    char* p = ws + off;
    off += (bytes + 255) & ~(size_t)255;
    return p;
  };
  bf16*  xT   = (bf16*)take((size_t)B_ * N_ * C_ * 2);
  bf16*  Wqb  = (bf16*)take((size_t)CQK_ * C_ * 2);
  bf16*  Wkb  = (bf16*)take((size_t)CQK_ * C_ * 2);
  bf16*  Wvb  = (bf16*)take((size_t)C_ * C_ * 2);
  bf16*  QtB  = (bf16*)take((size_t)B_ * N_ * CQK_ * 2);
  bf16*  KtB  = (bf16*)take((size_t)B_ * N_ * CQK_ * 2);
  bf16*  Vb   = (bf16*)take((size_t)B_ * C_ * N_ * 2);
  bf16*  Vs   = (bf16*)take((size_t)B_ * C_ * N_ * 2);
  float* mRow = (float*)take((size_t)B_ * N_ * 4);
  float* zinv = (float*)take((size_t)B_ * N_ * 4);
  (void)in_sizes; (void)n_in; (void)out_size; (void)ws_size;

  cvt_bf16_kernel<<<(CQK_ * C_ + 255) / 256, 256, 0, stream>>>(Wq, Wqb, CQK_ * C_);
  cvt_bf16_kernel<<<(CQK_ * C_ + 255) / 256, 256, 0, stream>>>(Wk, Wkb, CQK_ * C_);
  cvt_bf16_kernel<<<(C_ * C_ + 255) / 256, 256, 0, stream>>>(Wv, Wvb, C_ * C_);
  transpose_x_kernel<<<dim3(N_ / 32, C_ / 32, B_), dim3(32, 8), 0, stream>>>(x, xT);
  proj_kernel<<<(B_ * (CQK_ / 16) * (N_ / 16)) / 8, 256, 0, stream>>>(Wqb, bq, xT, QtB, CQK_, 1);
  proj_kernel<<<(B_ * (CQK_ / 16) * (N_ / 16)) / 8, 256, 0, stream>>>(Wkb, bk, xT, KtB, CQK_, 1);
  proj_kernel<<<(B_ * (C_ / 16) * (N_ / 16)) / 8, 256, 0, stream>>>(Wvb, bv, xT, Vb, C_, 0);
  stats_kernel<<<B_ * (N_ / 16), 128, 0, stream>>>(QtB, KtB, mRow, zinv);
  scalev_kernel<<<(int)(((size_t)B_ * C_ * N_ + 255) / 256), 256, 0, stream>>>(Vb, zinv, Vs);
  attn_out_kernel<<<B_ * (N_ / 64), 512, 0, stream>>>(QtB, KtB, Vs, mRow, x, gamma, out);
}
